// EpisodicMemoryBank_57088705298854
// MI455X (gfx1250) — matrix operand order repaired and checked by reading, not compiled, not run on hardware
//
#include <hip/hip_runtime.h>
#include <hip/hip_bf16.h>
#include <climits>

typedef __attribute__((ext_vector_type(16))) __bf16        v16bf;
typedef __attribute__((ext_vector_type(8)))  float         v8f;
typedef __attribute__((ext_vector_type(4)))  unsigned int  v4u;
typedef __attribute__((ext_vector_type(8)))  unsigned int  v8u;

#define B_  256
#define S_  1024
#define KD_ 128
#define VD_ 256
#define ED_ 128
#define CB_ 1024
#define IN_ 2144

// ---------------------------------------------------------------------------
// Pack wi = concat[write_summary|state|action] -> contiguous bf16 [256, 2144].
// ---------------------------------------------------------------------------
__global__ __launch_bounds__(256)
void emb_pack_wi(const float* __restrict__ wsum, const float* __restrict__ cstate,
                 const float* __restrict__ actsum, __bf16* __restrict__ out)
{
    int idx = blockIdx.x * 256 + threadIdx.x;          // B_*IN_ = 548864
    if (idx >= B_ * IN_) return;
    int b = idx / IN_, k = idx - b * IN_;
    float v = (k < 2048) ? wsum[b * 2048 + k]
            : (k < 2112) ? cstate[b * 64 + (k - 2048)]
                         : actsum[b * 32 + (k - 2112)];
    out[idx] = (__bf16)v;
}

// ---------------------------------------------------------------------------
// Pack + transpose weights: W[K][N] f32 -> Wt[N][K/2] u32, each u32 holding
// bf16(W[2j][n]) in the low half and bf16(W[2j+1][n]) in the high half —
// exactly the WMMA B-fragment bit order, so the GEMM can load fragments
// straight from global memory with b128 loads (no LDS, no in-loop transpose).
// ---------------------------------------------------------------------------
__global__ __launch_bounds__(256)
void emb_pack_wt(const float* __restrict__ W, unsigned int* __restrict__ Wt,
                 int K, int N)
{
    int K2 = K >> 1;
    int idx = blockIdx.x * 256 + threadIdx.x;          // N * K2 outputs
    if (idx >= N * K2) return;
    int n = idx / K2, j = idx - n * K2;
    float w0 = W[(size_t)(2 * j)     * N + n];
    float w1 = W[(size_t)(2 * j + 1) * N + n];
    union { unsigned int u; __bf16 h[2]; } pk;
    pk.h[0] = (__bf16)w0; pk.h[1] = (__bf16)w1;
    Wt[idx] = pk.u;
}

// ---------------------------------------------------------------------------
// WMMA bf16 GEMM, LDS-free:  C[M,N] = act( A[M,K](bf16) @ W[K,N] + bias )
// A is bf16 row-major; W is pre-packed/transposed Wt[N][K/2] (see above).
// Block = 128 threads (4 waves), tile = 16(M) x 128(N); each wave owns two
// 16x16 N-subtiles. Per K-step each lane issues 6 global b128 loads (2 for
// the shared A fragment, 4 for the two B fragments) + 2 WMMAs. Uniform
// control flow keeps EXEC all-ones as WMMA requires.
// ---------------------------------------------------------------------------
__global__ __launch_bounds__(128)
void emb_gemm_wmma(const __bf16* __restrict__ A, const unsigned int* __restrict__ Wt,
                   const float* __restrict__ bias,
                   float* __restrict__ Cf, __bf16* __restrict__ Cb,
                   int N, int K, int act)
{
    const int tid  = threadIdx.x;
    const int wv   = tid >> 5;
    const int lane = tid & 31;
    const int hl   = lane >> 4;
    const int l16  = lane & 15;
    const int mBase = blockIdx.y * 16;
    const int nBase = blockIdx.x * 128;
    const int K2   = K >> 1;

    const int c0 = nBase + wv * 16 + l16;   // subtile-0 column
    const int c1 = c0 + 64;                 // subtile-1 column

    // fragment-resident base pointers (advance by the K-step each iteration)
    const __bf16*       ap  = A  + (size_t)(mBase + l16) * K + hl * 16;
    const unsigned int* b0p = Wt + (size_t)c0 * K2 + hl * 8;
    const unsigned int* b1p = Wt + (size_t)c1 * K2 + hl * 8;

    v8f acc0 = {0.f,0.f,0.f,0.f,0.f,0.f,0.f,0.f};
    v8f acc1 = acc0;

    for (int kt = 0; kt < K; kt += 32) {
        v4u a0 = *(const v4u*)ap;            // k = kt + hl*16 + 0..7
        v4u a1 = *(const v4u*)(ap + 8);      // k = kt + hl*16 + 8..15
        v4u b0 = *(const v4u*)b0p;           // k = kt + hl*16 + 0..7
        v4u b1 = *(const v4u*)(b0p + 4);     // k = kt + hl*16 + 8..15
        v4u b2 = *(const v4u*)b1p;
        v4u b3 = *(const v4u*)(b1p + 4);
        ap  += 32;
        b0p += 16;
        b1p += 16;

        v8u au; au[0]=a0[0]; au[1]=a0[1]; au[2]=a0[2]; au[3]=a0[3];
                au[4]=a1[0]; au[5]=a1[1]; au[6]=a1[2]; au[7]=a1[3];
        v16bf a = __builtin_bit_cast(v16bf, au);

        v8u bu; bu[0]=b0[0]; bu[1]=b0[1]; bu[2]=b0[2]; bu[3]=b0[3];
                bu[4]=b1[0]; bu[5]=b1[1]; bu[6]=b1[2]; bu[7]=b1[3];
        v16bf bA = __builtin_bit_cast(v16bf, bu);

        v8u cu; cu[0]=b2[0]; cu[1]=b2[1]; cu[2]=b2[2]; cu[3]=b2[3];
                cu[4]=b3[0]; cu[5]=b3[1]; cu[6]=b3[2]; cu[7]=b3[3];
        v16bf bB = __builtin_bit_cast(v16bf, cu);

        acc0 = __builtin_amdgcn_wmma_f32_16x16x32_bf16(false, a, false, bA, (short)0, acc0, false, false);
        acc1 = __builtin_amdgcn_wmma_f32_16x16x32_bf16(false, a, false, bB, (short)0, acc1, false, false);
    }

    // ---- epilogue: bias (+SiLU), store both subtiles ----
    #pragma unroll
    for (int half2 = 0; half2 < 2; ++half2) {
        const int col = nBase + half2 * 64 + wv * 16 + l16;
        const float bb = bias[col];
        v8f* accp = half2 ? &acc1 : &acc0;
        #pragma unroll
        for (int r = 0; r < 8; ++r) {
            int row = mBase + hl * 8 + r;    // C/D: VGPR r -> M = 8*half + r
            float v = (*accp)[r] + bb;
            if (act) v = v / (1.f + __expf(-v));       // SiLU
            if (Cf) Cf[(size_t)row * N + col] = v;
            if (Cb) Cb[(size_t)row * N + col] = (__bf16)v;
        }
    }
}

// ---------------------------------------------------------------------------
__global__ __launch_bounds__(128)
void emb_normalize_rows(const float* __restrict__ in, float* __restrict__ out)
{
    __shared__ float red[128];
    int b = blockIdx.x, t = threadIdx.x;
    float v = in[b * KD_ + t];
    red[t] = v * v;
    __syncthreads();
    for (int s = 64; s > 0; s >>= 1) { if (t < s) red[t] += red[t + s]; __syncthreads(); }
    float inv = 1.f / fmaxf(sqrtf(red[0]), 1e-6f);
    out[b * KD_ + t] = v * inv;
}

// ---------------------------------------------------------------------------
__global__ __launch_bounds__(256)
void emb_vq_argmin(const float* __restrict__ proj, const float* __restrict__ cb,
                   int* __restrict__ phase, float* __restrict__ vq_acc)
{
    __shared__ float p[ED_];
    __shared__ float bd[256];
    __shared__ int   bi[256];
    int b = blockIdx.x, t = threadIdx.x;
    if (t < ED_) p[t] = proj[b * ED_ + t];
    __syncthreads();
    float bestd = 3.4e38f; int besti = 0;
    for (int j = t; j < CB_; j += 256) {
        const float4* c = (const float4*)(cb + (size_t)j * ED_);
        float d = 0.f;
        #pragma unroll 8
        for (int k = 0; k < ED_ / 4; ++k) {
            float4 cv = c[k];
            float d0 = p[4*k]   - cv.x, d1 = p[4*k+1] - cv.y;
            float d2 = p[4*k+2] - cv.z, d3 = p[4*k+3] - cv.w;
            d += d0*d0 + d1*d1 + d2*d2 + d3*d3;
        }
        if (d < bestd) { bestd = d; besti = j; }
    }
    bd[t] = bestd; bi[t] = besti;
    __syncthreads();
    for (int s = 128; s > 0; s >>= 1) {
        if (t < s) {
            if (bd[t+s] < bd[t] || (bd[t+s] == bd[t] && bi[t+s] < bi[t])) {
                bd[t] = bd[t+s]; bi[t] = bi[t+s];
            }
        }
        __syncthreads();
    }
    if (t == 0) { phase[b] = bi[0]; atomicAdd(vq_acc, bd[0]); }
}

__global__ void emb_zero_acc(float* acc) { acc[0] = 0.f; }

__global__ void emb_vq_finalize(const float* __restrict__ acc, float* __restrict__ out)
{
    // vq_loss = cb_loss + 0.25*commitment = 1.25 * mean(min ||p-c||^2)
    out[0] = acc[0] * (1.25f / (float)(B_ * ED_));
}

// ---------------------------------------------------------------------------
__global__ __launch_bounds__(256)
void emb_slot_select(const float* __restrict__ wkey, const float* __restrict__ keys,
                     const unsigned char* __restrict__ filled,
                     int* __restrict__ slot)
{
    __shared__ float wk[KD_];
    __shared__ float mv[256]; __shared__ int mi[256];
    __shared__ int   fe[256]; __shared__ int hf[256]; __shared__ int he[256];
    int b = blockIdx.x, t = threadIdx.x;
    if (t < KD_) wk[t] = wkey[b * KD_ + t];
    __syncthreads();
    float bestv = -3.4e38f; int besti = 0;
    int firstE = INT_MAX, hasF = 0, hasE = 0;
    for (int s = t; s < S_; s += 256) {
        const float4* kr = (const float4*)(keys + ((size_t)b * S_ + s) * KD_);
        float d = 0.f;
        #pragma unroll 8
        for (int k = 0; k < KD_ / 4; ++k) {
            float4 kv = kr[k];
            d += wk[4*k]*kv.x + wk[4*k+1]*kv.y + wk[4*k+2]*kv.z + wk[4*k+3]*kv.w;
        }
        int f = filled[(size_t)b * S_ + s] != 0;
        float lg = f ? d : -10000.0f;
        if (lg > bestv) { bestv = lg; besti = s; }   // s ascending -> first max
        if (f) hasF = 1; else { hasE = 1; if (s < firstE) firstE = s; }
    }
    mv[t] = bestv; mi[t] = besti; fe[t] = firstE; hf[t] = hasF; he[t] = hasE;
    __syncthreads();
    for (int st = 128; st > 0; st >>= 1) {
        if (t < st) {
            if (mv[t+st] > mv[t] || (mv[t+st] == mv[t] && mi[t+st] < mi[t])) {
                mv[t] = mv[t+st]; mi[t] = mi[t+st];
            }
            fe[t] = min(fe[t], fe[t+st]);
            hf[t] |= hf[t+st]; he[t] |= he[t+st];
        }
        __syncthreads();
    }
    if (t == 0) {
        int firstEmpty = (fe[0] == INT_MAX) ? 0 : fe[0];
        int useEmpty = (!hf[0]) || ((mv[0] < 0.5f) && he[0]);
        slot[b] = useEmpty ? firstEmpty : mi[0];
    }
}

// ---------------------------------------------------------------------------
__global__ __launch_bounds__(256)
void emb_compute_upd(const float* __restrict__ keys, const float* __restrict__ values,
                     const float* __restrict__ wkey, const float* __restrict__ wval,
                     const unsigned char* __restrict__ filled,
                     const unsigned char* __restrict__ vmask,
                     const int* __restrict__ slot,
                     float* __restrict__ upd_k, float* __restrict__ upd_v)
{
    __shared__ float red[128];
    __shared__ float kc[128];
    int b = blockIdx.x, t = threadIdx.x;
    int sl = slot[b];
    int vm = vmask[b] != 0;
    int selF = vm && (filled[(size_t)b * S_ + sl] != 0);
    if (t < KD_) {
        float sk  = keys[((size_t)b * S_ + sl) * KD_ + t];
        float wkv = wkey[b * KD_ + t];
        float merged = 0.9f * sk + 0.1f * wkv;
        kc[t]  = selF ? merged : wkv;
        red[t] = selF ? merged * merged : 0.f;
    }
    __syncthreads();
    for (int s = 64; s > 0; s >>= 1) { if (t < s) red[t] += red[t + s]; __syncthreads(); }
    if (t < KD_) {
        float o = kc[t];
        if (selF) o = o / fmaxf(sqrtf(red[0]), 1e-6f);
        upd_k[b * KD_ + t] = o;
    }
    float sv  = values[((size_t)b * S_ + sl) * VD_ + t];
    float wvv = wval[b * VD_ + t];
    upd_v[b * VD_ + t] = selF ? (0.9f * sv + 0.1f * wvv) : wvv;
}

// ---------------------------------------------------------------------------
__global__ void emb_copy_keys(const float4* __restrict__ in, float4* __restrict__ out,
                              const float* __restrict__ upd_k, const int* __restrict__ slot,
                              const unsigned char* __restrict__ vmask, long long total4)
{
    long long i = (long long)blockIdx.x * blockDim.x + threadIdx.x;
    long long stride = (long long)gridDim.x * blockDim.x;
    for (; i < total4; i += stride) {
        long long e = i * 4;
        int b = (int)(e >> 17);              // S*KD = 131072
        int s = (int)((e >> 7) & (S_ - 1));  // KD = 128
        float4 v;
        if (s == slot[b] && vmask[b])
            v = *(const float4*)(upd_k + b * KD_ + (int)(e & (KD_ - 1)));
        else
            v = in[i];
        out[i] = v;
    }
}

__global__ void emb_copy_values(const float4* __restrict__ in, float4* __restrict__ out,
                                const float* __restrict__ upd_v, const int* __restrict__ slot,
                                const unsigned char* __restrict__ vmask, long long total4)
{
    long long i = (long long)blockIdx.x * blockDim.x + threadIdx.x;
    long long stride = (long long)gridDim.x * blockDim.x;
    for (; i < total4; i += stride) {
        long long e = i * 4;
        int b = (int)(e >> 18);              // S*VD = 262144
        int s = (int)((e >> 8) & (S_ - 1));  // VD = 256
        float4 v;
        if (s == slot[b] && vmask[b])
            v = *(const float4*)(upd_v + b * VD_ + (int)(e & (VD_ - 1)));
        else
            v = in[i];
        out[i] = v;
    }
}

// ---------------------------------------------------------------------------
__global__ __launch_bounds__(256)
void emb_meta_update(const int* __restrict__ phase_ids, const int* __restrict__ ages,
                     const float* __restrict__ conf, const unsigned char* __restrict__ filled,
                     const unsigned char* __restrict__ vmask, const int* __restrict__ slot,
                     const int* __restrict__ wr_phase,
                     float* __restrict__ o_phase, float* __restrict__ o_ages,
                     float* __restrict__ o_conf, float* __restrict__ o_filled)
{
    int idx = blockIdx.x * 256 + threadIdx.x;
    if (idx >= B_ * S_) return;
    int b = idx >> 10, s = idx & (S_ - 1);
    int vm = vmask[b] != 0;
    int isSel = vm && (s == slot[b]);
    int f = filled[idx] != 0;

    float c = conf[idx] * 0.99f;
    if (isSel) c = f ? fminf(conf[idx] + 0.5f, 1.f) : 1.f;
    o_conf[idx]   = c;
    o_phase[idx]  = (float)(isSel ? wr_phase[b] : phase_ids[idx]);
    o_ages[idx]   = (float)(isSel ? 0 : (ages[idx] + vm));
    o_filled[idx] = (f || isSel) ? 1.f : 0.f;
}

// ---------------------------------------------------------------------------
extern "C" void kernel_launch(void* const* d_in, const int* in_sizes, int n_in,
                              void* d_out, int out_size, void* d_ws, size_t ws_size,
                              hipStream_t stream)
{
    (void)in_sizes; (void)n_in; (void)out_size; (void)ws_size;

    const float* keys      = (const float*)d_in[0];
    const float* values    = (const float*)d_in[1];
    const int*   phase_ids = (const int*)d_in[2];
    const int*   ages      = (const int*)d_in[3];
    const float* conf      = (const float*)d_in[4];
    const unsigned char* filled = (const unsigned char*)d_in[5];
    const float* wsum   = (const float*)d_in[6];
    const float* cstate = (const float*)d_in[7];
    const float* actsum = (const float*)d_in[8];
    const unsigned char* vmask = (const unsigned char*)d_in[9];
    const float* Wk1 = (const float*)d_in[10]; const float* bk1 = (const float*)d_in[11];
    const float* Wk2 = (const float*)d_in[12]; const float* bk2 = (const float*)d_in[13];
    const float* Wv1 = (const float*)d_in[14]; const float* bv1 = (const float*)d_in[15];
    const float* Wv2 = (const float*)d_in[16]; const float* bv2 = (const float*)d_in[17];
    const float* Wp  = (const float*)d_in[18]; const float* bp  = (const float*)d_in[19];
    const float* cb  = (const float*)d_in[20];

    float* out = (float*)d_out;
    const long long O_KEYS  = 0;
    const long long O_VALS  = 33554432LL;
    const long long O_PHASE = 100663296LL;
    const long long O_AGES  = 100925440LL;
    const long long O_CONF  = 101187584LL;
    const long long O_FILL  = 101449728LL;
    const long long O_WKEY  = 101711872LL;
    const long long O_WVAL  = 101744640LL;
    const long long O_VQ    = 101810176LL;

    // workspace layout (float slots)
    float*        ws      = (float*)d_ws;
    __bf16*       Hk_bf   = (__bf16*)(ws);              // 262144 slots
    __bf16*       Hv_bf   = (__bf16*)(ws + 262144);     // 262144 slots
    __bf16*       wi_bf   = (__bf16*)(ws + 524288);     // 274432 slots
    __bf16*       wval_bf = (__bf16*)(ws + 798720);     // 32768 slots
    float*        wkraw   = ws + 831488;                // 32768
    float*        proj    = ws + 864256;                // 32768
    float*        updk    = ws + 897024;                // 32768
    float*        updv    = ws + 929792;                // 65536
    float*        vqacc   = ws + 995328;                // 1 (+pad)
    int*          phase   = (int*)(ws + 995332);        // 256
    int*          slot    = (int*)(ws + 995588);        // 256
    unsigned int* Wbig    = (unsigned int*)(ws + 995848);   // 2048*1072 = 2195456 slots (reused)
    unsigned int* Wk2t    = (unsigned int*)(ws + 3191304);  // 128*1024  = 131072
    unsigned int* Wv2t    = (unsigned int*)(ws + 3322376);  // 256*1024  = 262144
    unsigned int* Wpt     = (unsigned int*)(ws + 3584520);  // 128*128   = 16384
    // total ~3600904 slots ~= 14.4 MB

    emb_zero_acc<<<1, 1, 0, stream>>>(vqacc);
    emb_pack_wi<<<(B_ * IN_ + 255) / 256, 256, 0, stream>>>(wsum, cstate, actsum, wi_bf);

    const dim3 blk(128);
    // MLP layer 1 (K=2144, N=2048), SiLU, H written directly as bf16.
    // Wbig is packed, consumed, then repacked (stream order makes this safe).
    emb_pack_wt<<<(2048 * 1072 + 255) / 256, 256, 0, stream>>>(Wk1, Wbig, IN_, 2048);
    emb_gemm_wmma<<<dim3(16, 16), blk, 0, stream>>>(wi_bf, Wbig, bk1, nullptr, Hk_bf, 2048, IN_, 1);
    emb_pack_wt<<<(2048 * 1072 + 255) / 256, 256, 0, stream>>>(Wv1, Wbig, IN_, 2048);
    emb_gemm_wmma<<<dim3(16, 16), blk, 0, stream>>>(wi_bf, Wbig, bv1, nullptr, Hv_bf, 2048, IN_, 1);
    // MLP layer 2
    emb_pack_wt<<<(128 * 1024 + 255) / 256, 256, 0, stream>>>(Wk2, Wk2t, 2048, 128);
    emb_gemm_wmma<<<dim3(1, 16), blk, 0, stream>>>(Hk_bf, Wk2t, bk2, wkraw, nullptr, 128, 2048, 0);
    emb_pack_wt<<<(256 * 1024 + 255) / 256, 256, 0, stream>>>(Wv2, Wv2t, 2048, 256);
    emb_gemm_wmma<<<dim3(2, 16), blk, 0, stream>>>(Hv_bf, Wv2t, bv2, out + O_WVAL, wval_bf, 256, 2048, 0);
    emb_normalize_rows<<<256, 128, 0, stream>>>(wkraw, out + O_WKEY);
    // VQ projection (K=256, N=128)
    emb_pack_wt<<<(128 * 128 + 255) / 256, 256, 0, stream>>>(Wp, Wpt, 256, 128);
    emb_gemm_wmma<<<dim3(1, 16), blk, 0, stream>>>(wval_bf, Wpt, bp, proj, nullptr, 128, 256, 0);

    emb_vq_argmin<<<256, 256, 0, stream>>>(proj, cb, phase, vqacc);
    emb_vq_finalize<<<1, 1, 0, stream>>>(vqacc, out + O_VQ);

    emb_slot_select<<<256, 256, 0, stream>>>(out + O_WKEY, keys, filled, slot);
    emb_compute_upd<<<256, 256, 0, stream>>>(keys, values, out + O_WKEY, out + O_WVAL,
                                             filled, vmask, slot, updk, updv);

    emb_copy_keys<<<32768, 256, 0, stream>>>((const float4*)keys, (float4*)(out + O_KEYS),
                                             updk, slot, vmask, 8388608LL);
    emb_copy_values<<<65536, 256, 0, stream>>>((const float4*)values, (float4*)(out + O_VALS),
                                               updv, slot, vmask, 16777216LL);
    emb_meta_update<<<1024, 256, 0, stream>>>(phase_ids, ages, conf, filled, vmask, slot, phase,
                                              out + O_PHASE, out + O_AGES,
                                              out + O_CONF, out + O_FILL);
}
